// Decoder_18537078850210
// MI455X (gfx1250) — matrix operand order repaired and checked by reading, not compiled, not run on hardware
//
#include <hip/hip_runtime.h>
#include <hip/hip_bf16.h>
#include <stdint.h>

// Problem constants (from reference)
#define Bx 64
#define Tx 128
#define Dx 1024
#define Ex 256
#define SFx 128
#define IDx 64
#define DXx 32
#define KCAT 3328   // E + 2D (lstm_Wi cols) + D (lstm_Wh cols)
#define N4D  4096   // 4*D

// d_out layout (floats): concated0 | dc_logits | id_logits | before_sf | sf_logits
#define OFF_C0 0
#define OFF_DC 131072      // 64*2048
#define OFF_ID 133120      // + 64*32
#define OFF_BS 137216      // + 64*64
#define OFF_SF 16914432    // + 64*128*2048

typedef __attribute__((ext_vector_type(16))) __bf16 v16bf;
typedef __attribute__((ext_vector_type(8)))  __bf16 bf16x8;
typedef __attribute__((ext_vector_type(8)))  float  v8f;

union AFrag { v16bf v; bf16x8 h[2]; };

__device__ inline __bf16 f2bf(float f) {
  union { float f; unsigned u; } in; in.f = f;
  unsigned u = in.u;
  unsigned r = u + 0x7FFFu + ((u >> 16) & 1u);   // round-to-nearest-even
  union { unsigned short s; __bf16 b; } out;
  out.s = (unsigned short)(r >> 16);
  return out.b;
}

// ---------------------------------------------------------------------------
// bf16 WMMA GEMM:  C[M x N] = A[M x K] * B[N x K]^T + bias[N]
// block = 128 threads (4 waves). grid.x = N/128, grid.y = M/64.
// Each wave: two 16-col N tiles x four 16-row M tiles (64x32 of C).
// K loop unrolled by 64 with ping-pong fragment sets: each half prefetches
// the other set, so no loop-carried register copies and loads overlap WMMA.
// Requires K % 64 == 0 (true for all call sites: 3328, 2048, 1024).
// ---------------------------------------------------------------------------
__global__ __launch_bounds__(128) void gemm_bf16_wmma(
    const __bf16* __restrict__ A, int lda,
    const __bf16* __restrict__ Bm, int ldb,
    const float* __restrict__ bias,
    float* __restrict__ C, int ldc, int K)
{
  const int wave = threadIdx.x >> 5;
  const int lane = threadIdx.x & 31;
  const int lh   = lane & 15;     // N (B/C) or M (A) within tile
  const int hh   = lane >> 4;     // lane-half select
  const int n0   = blockIdx.x * 128 + wave * 32;   // two N tiles: n0, n0+16
  const int mB   = blockIdx.y * 64;

  v8f acc[4][2] = {};

  const __bf16* __restrict__ bRow0 = Bm + (size_t)(n0 + lh) * ldb + (hh << 4);
  const __bf16* __restrict__ bRow1 = bRow0 + (size_t)16 * ldb;
  const __bf16* __restrict__ aBase = A + (size_t)(mB + lh) * lda + hh * 16;

  v16bf bf0a, bf1a, bf0b, bf1b;
  AFrag afa[4], afb[4];

  // prologue: set-A <- K-block 0
  bf0a = *(const v16bf*)(bRow0);
  bf1a = *(const v16bf*)(bRow1);
#pragma unroll
  for (int i = 0; i < 4; ++i) {
    const __bf16* ap = aBase + (size_t)(i * 16) * lda;
    afa[i].h[0] = *(const bf16x8*)(ap);        // K = 16h + 0..7
    afa[i].h[1] = *(const bf16x8*)(ap + 8);    // K = 16h + 8..15
  }

  for (int kb = 0; kb < K; kb += 64) {
    // prefetch set-B <- K-block kb+32 (always in range: K % 64 == 0)
    bf0b = *(const v16bf*)(bRow0 + kb + 32);
    bf1b = *(const v16bf*)(bRow1 + kb + 32);
#pragma unroll
    for (int i = 0; i < 4; ++i) {
      const __bf16* ap = aBase + (size_t)(i * 16) * lda + kb + 32;
      afb[i].h[0] = *(const bf16x8*)(ap);
      afb[i].h[1] = *(const bf16x8*)(ap + 8);
    }
    // consume set-A
#pragma unroll
    for (int i = 0; i < 4; ++i) {
      acc[i][0] = __builtin_amdgcn_wmma_f32_16x16x32_bf16(
          false, afa[i].v, false, bf0a, (short)0, acc[i][0], false, false);
      acc[i][1] = __builtin_amdgcn_wmma_f32_16x16x32_bf16(
          false, afa[i].v, false, bf1a, (short)0, acc[i][1], false, false);
    }
    // prefetch set-A <- K-block kb+64 (clamped on final iteration)
    const int kn = (kb + 64 < K) ? (kb + 64) : (kb + 32);
    bf0a = *(const v16bf*)(bRow0 + kn);
    bf1a = *(const v16bf*)(bRow1 + kn);
#pragma unroll
    for (int i = 0; i < 4; ++i) {
      const __bf16* ap = aBase + (size_t)(i * 16) * lda + kn;
      afa[i].h[0] = *(const bf16x8*)(ap);
      afa[i].h[1] = *(const bf16x8*)(ap + 8);
    }
    // consume set-B
#pragma unroll
    for (int i = 0; i < 4; ++i) {
      acc[i][0] = __builtin_amdgcn_wmma_f32_16x16x32_bf16(
          false, afb[i].v, false, bf0b, (short)0, acc[i][0], false, false);
      acc[i][1] = __builtin_amdgcn_wmma_f32_16x16x32_bf16(
          false, afb[i].v, false, bf1b, (short)0, acc[i][1], false, false);
    }
  }

#pragma unroll
  for (int j = 0; j < 2; ++j) {
    const int col = n0 + j * 16 + lh;
    const float bv = bias ? bias[col] : 0.0f;
#pragma unroll
    for (int i = 0; i < 4; ++i) {
      const int row = mB + i * 16 + hh * 8;
#pragma unroll
      for (int r = 0; r < 8; ++r)
        C[(size_t)(row + r) * ldc + col] = acc[i][j][r] + bv;
    }
  }
}

// ---------------------------------------------------------------------------
// One-time conversion / assembly kernels
// ---------------------------------------------------------------------------
__global__ void k_f2bf(const float* __restrict__ src, __bf16* __restrict__ dst, int n) {
  int i = blockIdx.x * blockDim.x + threadIdx.x;
  if (i < n) dst[i] = f2bf(src[i]);
}

__global__ void k_build_wcat(const float* __restrict__ Wi, const float* __restrict__ Wh,
                             __bf16* __restrict__ Wcat) {
  size_t idx = (size_t)blockIdx.x * blockDim.x + threadIdx.x;
  if (idx >= (size_t)N4D * KCAT) return;
  int n = (int)(idx / KCAT), k = (int)(idx % KCAT);
  float v = (k < 2304) ? Wi[(size_t)n * 2304 + k] : Wh[(size_t)n * Dx + (k - 2304)];
  Wcat[idx] = f2bf(v);
}

__global__ void k_bias4(const float* bi, const float* bh, float* bias4) {
  int i = blockIdx.x * blockDim.x + threadIdx.x;
  if (i < N4D) bias4[i] = bi[i] + bh[i];
}

// h0=c0=0, ctx0 = context[:,0,:], emb0 = emb_W[input_ids]
__global__ void k_init_state(const float* __restrict__ context, const int* __restrict__ ids,
                             const float* __restrict__ embW,
                             float* c, float* h, float* ctxf,
                             __bf16* X, __bf16* Xc) {
  int idx = blockIdx.x * blockDim.x + threadIdx.x;  // 64*1024
  int b = idx >> 10, d = idx & 1023;
  c[idx] = 0.f; h[idx] = 0.f;
  float cv = context[(size_t)b * Dx + d];
  ctxf[idx] = cv;
  X[(size_t)b * KCAT + 256 + d]  = f2bf(cv);
  X[(size_t)b * KCAT + 2304 + d] = f2bf(0.f);
  Xc[(size_t)b * 2048 + 1024 + d] = f2bf(cv);
  Xc[(size_t)b * 2048 + d]        = f2bf(0.f);
  if (d < Ex) X[(size_t)b * KCAT + d] = f2bf(embW[(size_t)ids[b] * Ex + d]);
}

// X[:,1280:2304] = enc_bf16[:, t, :]
__global__ void k_copy_aligned(const __bf16* __restrict__ encb, __bf16* __restrict__ X, int t) {
  int idx = blockIdx.x * blockDim.x + threadIdx.x;
  int b = idx >> 10, d = idx & 1023;
  X[(size_t)b * KCAT + 1280 + d] = encb[(size_t)b * Tx * Dx + (size_t)t * Dx + d];
}

// ---------------------------------------------------------------------------
// Per-step pointwise kernels
// ---------------------------------------------------------------------------
__global__ void k_lstm_pw(const float* __restrict__ gates, float* __restrict__ c,
                          float* __restrict__ h, const float* __restrict__ ctxf,
                          float* __restrict__ out_before, int t,
                          __bf16* __restrict__ X, __bf16* __restrict__ Xc) {
  int idx = blockIdx.x * blockDim.x + threadIdx.x;  // 64*1024
  int b = idx >> 10, d = idx & 1023;
  const float* g = gates + (size_t)b * N4D;
  float ig = g[d], fg = g[1024 + d], gg = g[2048 + d], og = g[3072 + d];
  float si = 1.f / (1.f + __expf(-ig));
  float sf = 1.f / (1.f + __expf(-fg));
  float so = 1.f / (1.f + __expf(-og));
  float cn = sf * c[idx] + si * tanhf(gg);
  float hn = so * tanhf(cn);
  c[idx] = cn; h[idx] = hn;
  X[(size_t)b * KCAT + 2304 + d] = f2bf(hn);
  Xc[(size_t)b * 2048 + d]       = f2bf(hn);
  float* ob = out_before + (size_t)b * Tx * 2048 + (size_t)t * 2048;
  ob[d] = hn;
  ob[1024 + d] = ctxf[idx];   // carried-in ctx
}

// sf_logits slice, argmax (first-max tie-break), gather next embedding
__global__ void k_slot_post(const float* __restrict__ score, float* __restrict__ out_sf,
                            int t, const __bf16* __restrict__ embW, __bf16* __restrict__ X) {
  int b = blockIdx.x, s = threadIdx.x;  // 128 threads
  float v = score[b * SFx + s];
  out_sf[(size_t)b * Tx * SFx + (size_t)t * SFx + s] = v;
  __shared__ float sv[SFx];
  __shared__ int   si[SFx];
  sv[s] = v; si[s] = s; __syncthreads();
  for (int o = 64; o; o >>= 1) {
    if (s < o) {
      if (sv[s + o] > sv[s] || (sv[s + o] == sv[s] && si[s + o] < si[s])) {
        sv[s] = sv[s + o]; si[s] = si[s + o];
      }
    }
    __syncthreads();
  }
  int nxt = si[0];
  const __bf16* er = embW + (size_t)nxt * Ex;
  X[(size_t)b * KCAT + s]       = er[s];
  X[(size_t)b * KCAT + 128 + s] = er[128 + s];
}

// scores[b][t] = dot(eW[b,t,:], h[b,:]) — one wave per (b,t)
__global__ void k_attn_scores(const float* __restrict__ eW, const float* __restrict__ h,
                              float* __restrict__ sc) {
  int gtid = blockIdx.x * blockDim.x + threadIdx.x;
  int wid = gtid >> 5, lane = threadIdx.x & 31;
  int b = wid >> 7, t = wid & (Tx - 1);
  const float* e  = eW + ((size_t)b * Tx + t) * Dx;
  const float* hb = h + (size_t)b * Dx;
  float s = 0.f;
  for (int d = lane; d < Dx; d += 32) s += e[d] * hb[d];
#pragma unroll
  for (int o = 16; o; o >>= 1) s += __shfl_xor(s, o, 32);
  if (lane == 0) sc[wid] = s;
}

__global__ void k_softmax_mask(const float* __restrict__ sc, const unsigned char* __restrict__ mask,
                               float* __restrict__ alpha) {
  int b = blockIdx.x, t = threadIdx.x;  // 128 threads
  __shared__ float red[Tx];
  float v = sc[b * Tx + t];
  if (mask[b * Tx + t]) v = -1e12f;
  red[t] = v; __syncthreads();
  for (int o = 64; o; o >>= 1) { if (t < o) red[t] = fmaxf(red[t], red[t + o]); __syncthreads(); }
  float m = red[0]; __syncthreads();
  float e = __expf(v - m);
  red[t] = e; __syncthreads();
  for (int o = 64; o; o >>= 1) { if (t < o) red[t] += red[t + o]; __syncthreads(); }
  alpha[b * Tx + t] = e / red[0];
}

// ctx[b][d] = sum_t alpha[b][t] * enc[b][t][d]   (coalesced over d)
__global__ void k_ctx(const float* __restrict__ alpha, const float* __restrict__ enc,
                      float* __restrict__ ctxf, __bf16* __restrict__ X,
                      __bf16* __restrict__ Xc, int writeX) {
  int b = blockIdx.x;
  int d = blockIdx.y * 256 + threadIdx.x;
  const float* e = enc + (size_t)b * Tx * Dx + d;
  const float* a = alpha + b * Tx;
  float s = 0.f;
  for (int t = 0; t < Tx; ++t) s += a[t] * e[(size_t)t * Dx];
  ctxf[(size_t)b * Dx + d] = s;
  if (writeX) {
    X[(size_t)b * KCAT + 256 + d]   = f2bf(s);
    Xc[(size_t)b * 2048 + 1024 + d] = f2bf(s);
  }
}

// concated0 + intent/domain logits (h0 == 0, so only ctx half of W matters)
__global__ void k_head(const float* __restrict__ clsctx,
                       const float* __restrict__ iW, const float* __restrict__ ib,
                       const float* __restrict__ dW, const float* __restrict__ db,
                       float* __restrict__ out) {
  int b = blockIdx.x, t = threadIdx.x;  // 128 threads
  float* c0 = out + OFF_C0 + (size_t)b * 2048;
  const float* cc = clsctx + (size_t)b * Dx;
  for (int j = t; j < Dx; j += 128) { c0[j] = 0.f; c0[1024 + j] = cc[j]; }
  if (t < IDx) {
    const float* w = iW + (size_t)t * 2048 + 1024;
    float s = ib[t];
    for (int k = 0; k < Dx; ++k) s += w[k] * cc[k];
    out[OFF_ID + b * IDx + t] = s;
  } else if (t < IDx + DXx) {
    int j = t - IDx;
    const float* w = dW + (size_t)j * 2048 + 1024;
    float s = db[j];
    for (int k = 0; k < Dx; ++k) s += w[k] * cc[k];
    out[OFF_DC + b * DXx + j] = s;
  }
}

// ---------------------------------------------------------------------------
extern "C" void kernel_launch(void* const* d_in, const int* in_sizes, int n_in,
                              void* d_out, int out_size, void* d_ws, size_t ws_size,
                              hipStream_t stream) {
  const int*   ids    = (const int*)d_in[0];
  const float* contex = (const float*)d_in[1];
  const float* enc    = (const float*)d_in[2];
  const unsigned char* mask = (const unsigned char*)d_in[3];
  const float* embW   = (const float*)d_in[5];
  const float* Wi     = (const float*)d_in[6];
  const float* Wh     = (const float*)d_in[7];
  const float* bi     = (const float*)d_in[8];
  const float* bh     = (const float*)d_in[9];
  const float* attnW  = (const float*)d_in[10];
  const float* attnB  = (const float*)d_in[11];
  const float* slotW  = (const float*)d_in[12];
  const float* slotB  = (const float*)d_in[13];
  const float* intW   = (const float*)d_in[14];
  const float* intB   = (const float*)d_in[15];
  const float* domW   = (const float*)d_in[16];
  const float* domB   = (const float*)d_in[17];
  float* out = (float*)d_out;

  // workspace carve-out (256B aligned)
  uintptr_t p = (uintptr_t)d_ws;
  auto alloc = [&](size_t bytes) -> void* {
    void* r = (void*)p; p += (bytes + 255) & ~(size_t)255; return r;
  };
  __bf16* encb   = (__bf16*)alloc((size_t)Bx * Tx * Dx * 2);
  __bf16* Wcat   = (__bf16*)alloc((size_t)N4D * KCAT * 2);
  __bf16* slotWb = (__bf16*)alloc((size_t)SFx * 2048 * 2);
  __bf16* attnWb = (__bf16*)alloc((size_t)Dx * Dx * 2);
  __bf16* embWb  = (__bf16*)alloc((size_t)SFx * Ex * 2);
  float*  eW     = (float*)alloc((size_t)Bx * Tx * Dx * 4);
  __bf16* X      = (__bf16*)alloc((size_t)Bx * KCAT * 2);
  __bf16* Xc     = (__bf16*)alloc((size_t)Bx * 2048 * 2);
  float*  gates  = (float*)alloc((size_t)Bx * N4D * 4);
  float*  cbuf   = (float*)alloc((size_t)Bx * Dx * 4);
  float*  hbuf   = (float*)alloc((size_t)Bx * Dx * 4);
  float*  ctxf   = (float*)alloc((size_t)Bx * Dx * 4);
  float*  clsctx = (float*)alloc((size_t)Bx * Dx * 4);
  float*  scr    = (float*)alloc((size_t)Bx * Tx * 4);
  float*  alpha  = (float*)alloc((size_t)Bx * Tx * 4);
  float*  score  = (float*)alloc((size_t)Bx * SFx * 4);
  float*  bias4  = (float*)alloc((size_t)N4D * 4);

  // --- one-time conversions -------------------------------------------------
  {
    int n = Bx * Tx * Dx;
    k_f2bf<<<(n + 255) / 256, 256, 0, stream>>>(enc, encb, n);
    n = Dx * Dx;
    k_f2bf<<<(n + 255) / 256, 256, 0, stream>>>(attnW, attnWb, n);
    n = SFx * 2048;
    k_f2bf<<<(n + 255) / 256, 256, 0, stream>>>(slotW, slotWb, n);
    n = SFx * Ex;
    k_f2bf<<<(n + 255) / 256, 256, 0, stream>>>(embW, embWb, n);
    size_t nw = (size_t)N4D * KCAT;
    k_build_wcat<<<(unsigned)((nw + 255) / 256), 256, 0, stream>>>(Wi, Wh, Wcat);
    k_bias4<<<N4D / 256, 256, 0, stream>>>(bi, bh, bias4);
    k_init_state<<<(Bx * Dx) / 256, 256, 0, stream>>>(contex, ids, embW, cbuf, hbuf, ctxf, X, Xc);
  }

  // eW = enc @ attn_W^T + attn_b   (M=B*T=8192, N=D=1024, K=D=1024)
  gemm_bf16_wmma<<<dim3(Dx / 128, (Bx * Tx) / 64), 128, 0, stream>>>(
      encb, Dx, attnWb, Dx, attnB, eW, Dx, Dx);

  // --- cls attention (h0 = 0) + heads --------------------------------------
  k_attn_scores<<<(Bx * Tx * 32) / 256, 256, 0, stream>>>(eW, hbuf, scr);
  k_softmax_mask<<<Bx, Tx, 0, stream>>>(scr, mask, alpha);
  k_ctx<<<dim3(Bx, Dx / 256), 256, 0, stream>>>(alpha, enc, clsctx, X, Xc, 0);
  k_head<<<Bx, 128, 0, stream>>>(clsctx, intW, intB, domW, domB, out);

  // --- sequential scan over T steps ----------------------------------------
  float* out_before = out + OFF_BS;
  float* out_sf     = out + OFF_SF;
  for (int t = 0; t < Tx; ++t) {
    k_copy_aligned<<<(Bx * Dx) / 256, 256, 0, stream>>>(encb, X, t);
    // gates = X @ Wcat^T + (bi+bh)   (M=64, N=4096, K=3328)
    gemm_bf16_wmma<<<dim3(N4D / 128, 1), 128, 0, stream>>>(
        X, KCAT, Wcat, KCAT, bias4, gates, N4D, KCAT);
    k_lstm_pw<<<(Bx * Dx) / 256, 256, 0, stream>>>(gates, cbuf, hbuf, ctxf,
                                                   out_before, t, X, Xc);
    // score = [h, ctx] @ slot_W^T + slot_b   (M=64, N=128, K=2048)
    gemm_bf16_wmma<<<dim3(SFx / 128, 1), 128, 0, stream>>>(
        Xc, 2048, slotWb, 2048, slotB, score, SFx, 2048);
    k_slot_post<<<Bx, SFx, 0, stream>>>(score, out_sf, t, embWb, X);
    // attention with new h -> ctx for next step
    k_attn_scores<<<(Bx * Tx * 32) / 256, 256, 0, stream>>>(eW, hbuf, scr);
    k_softmax_mask<<<Bx, Tx, 0, stream>>>(scr, mask, alpha);
    k_ctx<<<dim3(Bx, Dx / 256), 256, 0, stream>>>(alpha, enc, ctxf, X, Xc, 1);
  }
}
